// DimeNetPlusPlus_13142599926311
// MI455X (gfx1250) — compile-verified, matrix operand order checked by reading, not run on hardware
//
#include <hip/hip_runtime.h>

// DimeNet++ interaction block for gfx1250 (MI455X).
// bf16 WMMA (v_wmma_f32_16x16x32_bf16) + fp32 accumulate for all GEMMs,
// TDM (tensor_load_to_lds) for tile staging, f32 atomics for segment_sum.
// ~300 GFLOP GEMM / ~2.5-3 GB HBM traffic -> target the 23.3 TB/s floor.

#define H    192
#define IDIM 64
#define NRAD 6
#define NSBF 42
#define MT   64          // M rows per block (4 row-fragments per wave)
#define LSTR 200         // bf16 LDS row stride (100 dw -> 16 distinct banks)
#define XSTR 195         // f32 dw stride of TDM-padded x tile (192 + 3 pads)
#define GSTR 65          // f32 dw stride of TDM-padded agg tile (64 + 1 pad)
#define MS   (H * H)

typedef __attribute__((ext_vector_type(16))) __bf16 bf16x16;
typedef __attribute__((ext_vector_type(8)))  float  v8f;
typedef __attribute__((ext_vector_type(4)))  int    i32x4;
typedef __attribute__((ext_vector_type(8)))  int    i32x8;

static __device__ __forceinline__ float silu_f(float v) {
  return v / (1.0f + __expf(-v));
}

static __device__ __forceinline__ v8f wmma_bf16(bf16x16 a, bf16x16 b, v8f c) {
  return __builtin_amdgcn_wmma_f32_16x16x32_bf16(false, a, false, b, (short)0, c,
                                                 false, false);
}

// ---------------- Tensor Data Mover: 2-D tile load, ISA 08 D# layout --------
// group0: [1:0]=count=1, [63:32]=lds_addr, [120:64]=global_addr, [127:126]=2
// group1: [17:16]=data_size(2=4B), [20]=pad_en, [24:22]=pad_interval(5=64dw),
//         [31:25]=pad_amount(0=1dw), [79:48]=tensor_dim0, [111:80]=tensor_dim1,
//         [127:112]=tile_dim0, [143:128]=tile_dim1, [207:160]=dim0_stride
static __device__ __forceinline__ void tdm_load_2d_f32(
    unsigned lds_byte_off, const void* gptr, unsigned tensor_d0, unsigned tensor_d1,
    unsigned tile_d0, unsigned tile_d1, unsigned d0_stride, int pad_every64) {
  unsigned long long ga = (unsigned long long)gptr;
  i32x4 g0;
  g0[0] = 1;
  g0[1] = (int)lds_byte_off;
  g0[2] = (int)(unsigned)ga;
  g0[3] = (int)((((unsigned)(ga >> 32)) & 0x1ffffffu) | 0x80000000u);
  unsigned d0 = (2u << 16);
  if (pad_every64) d0 |= (1u << 20) | (5u << 22);   // pad 1 dword per 64 dwords
  i32x8 g1;
  g1[0] = (int)d0;
  g1[1] = (int)((tensor_d0 & 0xffffu) << 16);
  g1[2] = (int)((tensor_d0 >> 16) | ((tensor_d1 & 0xffffu) << 16));
  g1[3] = (int)((tensor_d1 >> 16) | (tile_d0 << 16));
  g1[4] = (int)(tile_d1 & 0xffffu);
  g1[5] = (int)d0_stride;
  g1[6] = 0;
  g1[7] = 0;
  asm volatile("tensor_load_to_lds %0, %1" :: "s"(g0), "s"(g1) : "memory");
}
static __device__ __forceinline__ void tdm_wait() {
  __builtin_amdgcn_s_wait_tensorcnt(0);
}

// ---------------- fragment helpers (ISA 7.12.2 16-bit layouts) --------------
// A 16x32: lane<16 -> M=lane, K={0..7,16..23}; lane>=16 -> M=lane-16, K+8.
static __device__ __forceinline__ bf16x16 load_a_frag(const __bf16* s, int stride,
                                                      int kbase, int lane) {
  const int row = lane & 15;
  const int kh  = (lane & 16) ? 8 : 0;
  union { bf16x16 v; unsigned u[8]; } r;
  const unsigned* base = (const unsigned*)(s + row * stride + kbase + kh);
  r.u[0] = base[0];  r.u[1] = base[1];  r.u[2] = base[2];  r.u[3] = base[3];
  r.u[4] = base[8];  r.u[5] = base[9];  r.u[6] = base[10]; r.u[7] = base[11];
  return r.v;
}

// A fragment from a TDM-padded f32 tile (1 pad dword per 64): off = k + k/64.
static __device__ __forceinline__ bf16x16 load_a_frag_f32(const float* s, int strideDw,
                                                          int row0, int kbase, int lane) {
  const int row = row0 + (lane & 15);
  const int kb  = kbase + ((lane & 16) ? 8 : 0);
  union { bf16x16 v; __bf16 h[16]; } r;
  const float* base = s + row * strideDw;
#pragma unroll
  for (int p = 0; p < 8; ++p) {
    int k = kb + ((p < 4) ? 2 * p : 2 * p + 8);
    int off = k + (k >> 6);
    r.h[2 * p]     = (__bf16)base[off];
    r.h[2 * p + 1] = (__bf16)base[off + 1];
  }
  return r.v;
}

// A fragment from unpadded f32 sbf tile (42 valid K, zero-fill to 64).
static __device__ __forceinline__ bf16x16 load_a_frag_sbf(const float* s, int row0,
                                                          int kbase, int lane) {
  const int row = row0 + (lane & 15);
  const int kb  = kbase + ((lane & 16) ? 8 : 0);
  union { bf16x16 v; __bf16 h[16]; } r;
  const float* base = s + row * NSBF;
#pragma unroll
  for (int p = 0; p < 8; ++p) {
    int k = kb + ((p < 4) ? 2 * p : 2 * p + 8);
    r.h[2 * p]     = (k < NSBF)     ? (__bf16)base[k]     : (__bf16)0.0f;
    r.h[2 * p + 1] = (k + 1 < NSBF) ? (__bf16)base[k + 1] : (__bf16)0.0f;
  }
  return r.v;
}

static __device__ __forceinline__ bf16x16 load_b_frag(const __bf16* w, int frag,
                                                      int lane) {
  return *(const bf16x16*)(w + (((size_t)frag * 32 + lane) << 4));
}

static __device__ __forceinline__ void store_frag(__bf16* s, v8f v, int row0,
                                                  int rbase, int col) {
#pragma unroll
  for (int r = 0; r < 8; ++r) s[(row0 + rbase + r) * LSTR + col] = (__bf16)v[r];
}

// silu(src @ W + b) for one wave's 64x16 tile (K=192); B frag reused 4x.
static __device__ __forceinline__ void layer192_m64(const __bf16* s,
                                                    const __bf16* __restrict__ Wp,
                                                    const float* __restrict__ bias,
                                                    int nt, int lane, int col,
                                                    v8f out[4]) {
  v8f acc[4] = {};
#pragma unroll
  for (int ks = 0; ks < 6; ++ks) {
    bf16x16 b = load_b_frag(Wp, ks * 12 + nt, lane);
#pragma unroll
    for (int rt = 0; rt < 4; ++rt) {
      bf16x16 a = load_a_frag(s + rt * 16 * LSTR, LSTR, ks * 32, lane);
      acc[rt] = wmma_bf16(a, b, acc[rt]);
    }
  }
  float bv = bias[col];
#pragma unroll
  for (int rt = 0; rt < 4; ++rt)
#pragma unroll
    for (int r = 0; r < 8; ++r) out[rt][r] = silu_f(acc[rt][r] + bv);
}

// ---------------------------------------------------------------------------
__global__ void zero_f32_kernel(float* __restrict__ p, long n) {
  long i = (long)blockIdx.x * blockDim.x + threadIdx.x;
  if (i < n) p[i] = 0.0f;
}

__global__ void combine_weights_kernel(const float* __restrict__ Wr1,
                                       const float* __restrict__ Wr2,
                                       const float* __restrict__ Ws1,
                                       const float* __restrict__ Ws2,
                                       float* __restrict__ Wrbf_c,
                                       float* __restrict__ Wsbf_c) {
  int tid = blockIdx.x * blockDim.x + threadIdx.x;
  if (tid < NRAD * H) {
    int j = tid / H, n = tid - j * H;
    float s = 0.0f;
#pragma unroll
    for (int b = 0; b < 8; ++b) s += Wr1[j * 8 + b] * Wr2[b * H + n];
    Wrbf_c[tid] = s;
  }
  int t2 = tid - NRAD * H;
  if (t2 >= 0 && t2 < NSBF * IDIM) {
    int k = t2 / IDIM, n = t2 - k * IDIM;
    float s = 0.0f;
#pragma unroll
    for (int b = 0; b < 8; ++b) s += Ws1[k * 8 + b] * Ws2[b * IDIM + n];
    Wsbf_c[t2] = s;
  }
}

__global__ void pack_weights_kernel(const float* __restrict__ src,
                                    __bf16* __restrict__ dst,
                                    int Ksrc, int Kpad, int N) {
  int tid = blockIdx.x * blockDim.x + threadIdx.x;
  if (tid >= Kpad * N) return;
  int e = tid & 15;
  int lane = (tid >> 4) & 31;
  int f = tid >> 9;
  int ntiles = N >> 4;
  int ntl = f % ntiles;
  int ks = f / ntiles;
  int k = ks * 32 + ((e < 8 ? e : e + 8) + ((lane & 16) ? 8 : 0));
  int n = ntl * 16 + (lane & 15);
  float v = (k < Ksrc) ? src[(size_t)k * N + n] : 0.0f;
  dst[tid] = (__bf16)v;
}

// ---------------------------------------------------------------------------
// Edge pre: x_ji = silu(x@Wji+b); t = silu(x@Wkj+b)*(rbf@Wrbf_c);
//           x_kj = silu(t@Wdown).  Block = 64 edges, 384 thr (12 waves).
__global__ __launch_bounds__(384) void edge_pre_kernel(
    const float* __restrict__ x, const float* __restrict__ rbf,
    const __bf16* __restrict__ Wji_p, const __bf16* __restrict__ Wkj_p,
    const __bf16* __restrict__ Wdown_p, const float* __restrict__ b_ji,
    const float* __restrict__ b_kj, const float* __restrict__ Wrbf_c,
    float* __restrict__ x_ji_out, float* __restrict__ xkj_out, int E) {
  __shared__ float  sX[MT * XSTR];    // TDM-padded f32 x tile
  __shared__ __bf16 sT[MT * LSTR];
  __shared__ float  sRbf[MT * 8];
  const int e0 = blockIdx.x * MT;
  const int tid = threadIdx.x, lane = tid & 31, wave = tid >> 5;

  if (wave == 0)
    tdm_load_2d_f32((unsigned)(size_t)sX, x + (size_t)e0 * H,
                    H, (unsigned)E, H, MT, H, 1);
  for (int i = tid; i < MT * NRAD; i += 384) {
    int r = i / NRAD, c = i - r * NRAD;
    int er = e0 + r;
    sRbf[r * 8 + c] = (er < E) ? rbf[(size_t)er * NRAD + c] : 0.0f;
  }
  if (wave == 0) tdm_wait();
  __syncthreads();

  const int nt = wave;                 // column tile 0..11
  const int nl = lane & 15;
  const int rbase = (lane & 16) ? 8 : 0;
  const int col = nt * 16 + nl;

  v8f accJ[4] = {}, accK[4] = {};
#pragma unroll
  for (int ks = 0; ks < 6; ++ks) {
    bf16x16 bj = load_b_frag(Wji_p, ks * 12 + nt, lane);
    bf16x16 bk = load_b_frag(Wkj_p, ks * 12 + nt, lane);
#pragma unroll
    for (int rt = 0; rt < 4; ++rt) {
      bf16x16 a = load_a_frag_f32(sX, XSTR, rt * 16, ks * 32, lane);
      accJ[rt] = wmma_bf16(a, bj, accJ[rt]);
      accK[rt] = wmma_bf16(a, bk, accK[rt]);
    }
  }
  const float bj = b_ji[col];
  const float bk = b_kj[col];
  float wr[NRAD];
#pragma unroll
  for (int j = 0; j < NRAD; ++j) wr[j] = Wrbf_c[j * H + col];

#pragma unroll
  for (int rt = 0; rt < 4; ++rt)
#pragma unroll
    for (int r = 0; r < 8; ++r) {
      int row = rt * 16 + rbase + r, er = e0 + row;
      float vj = silu_f(accJ[rt][r] + bj);
      if (er < E) x_ji_out[(size_t)er * H + col] = vj;
      float rh = 0.0f;
#pragma unroll
      for (int j = 0; j < NRAD; ++j) rh += sRbf[row * 8 + j] * wr[j];
      sT[row * LSTR + col] = (__bf16)(silu_f(accK[rt][r] + bk) * rh);
    }
  __syncthreads();

  if (wave < 8) {  // W_down: 4 col tiles x 2 row-halves over 8 waves
    const int ntd = wave & 3;
    const int rh0 = (wave >> 2) * 2;   // row-frag pair {rh0, rh0+1}
    v8f acc[2] = {};
#pragma unroll
    for (int ks = 0; ks < 6; ++ks) {
      bf16x16 b = load_b_frag(Wdown_p, ks * 4 + ntd, lane);
#pragma unroll
      for (int q = 0; q < 2; ++q) {
        bf16x16 a = load_a_frag(sT + (rh0 + q) * 16 * LSTR, LSTR, ks * 32, lane);
        acc[q] = wmma_bf16(a, b, acc[q]);
      }
    }
    int c2 = ntd * 16 + nl;
#pragma unroll
    for (int q = 0; q < 2; ++q)
#pragma unroll
      for (int r = 0; r < 8; ++r) {
        int er = e0 + (rh0 + q) * 16 + rbase + r;
        if (er < E) xkj_out[(size_t)er * IDIM + c2] = silu_f(acc[q][r]);
      }
  }
}

// ---------------------------------------------------------------------------
// Triplets: TDM-stage 256 sbf rows; sbf_i = sbf@Wsbf_c via WMMA (K 42->64);
// m = x_kj[idx_kj]*sbf_i; agg[idx_ji] += m (f32 atomics, agg L2-resident).
// Block = 256 thr (8 waves); wave does sub-tiles {w, w+8}; B frags in regs.
__global__ __launch_bounds__(256) void triplet_kernel(
    const float* __restrict__ sbf, const int* __restrict__ idx_kj,
    const int* __restrict__ idx_ji, const __bf16* __restrict__ Wsbf_p,
    const float* __restrict__ xkj, float* __restrict__ agg, long T) {
  __shared__ float sS[256 * NSBF];
  const int lane = threadIdx.x & 31;
  const int wave = threadIdx.x >> 5;
  const long t0 = (long)blockIdx.x * 256;

  if (wave == 0)
    tdm_load_2d_f32((unsigned)(size_t)sS, sbf + (size_t)t0 * NSBF,
                    NSBF, (unsigned)T, NSBF, 256, NSBF, 0);
  if (wave == 0) tdm_wait();
  __syncthreads();

  bf16x16 b0[4], b1[4];
#pragma unroll
  for (int nt = 0; nt < 4; ++nt) {
    b0[nt] = load_b_frag(Wsbf_p, nt, lane);
    b1[nt] = load_b_frag(Wsbf_p, 4 + nt, lane);
  }
  const int nl = lane & 15;
  const int rbase = (lane & 16) ? 8 : 0;

#pragma unroll
  for (int sub = 0; sub < 2; ++sub) {
    const int row0 = (wave + sub * 8) * 16;
    bf16x16 a0 = load_a_frag_sbf(sS, row0, 0, lane);
    bf16x16 a1 = load_a_frag_sbf(sS, row0, 32, lane);
    v8f acc[4] = {};
#pragma unroll
    for (int nt = 0; nt < 4; ++nt) {
      acc[nt] = wmma_bf16(a0, b0[nt], acc[nt]);
      acc[nt] = wmma_bf16(a1, b1[nt], acc[nt]);
    }
#pragma unroll
    for (int r = 0; r < 8; ++r) {
      long trip = t0 + row0 + rbase + r;
      if (trip < T) {
        int kj = idx_kj[trip];
        int ji = idx_ji[trip];
        const float* xr = xkj + (size_t)kj * IDIM;
        float* ar = agg + (size_t)ji * IDIM;
#pragma unroll
        for (int nt = 0; nt < 4; ++nt) {
          int c = nt * 16 + nl;
          atomicAdd(ar + c, acc[nt][r] * xr[c]);   // coalesced 64B per row
        }
      }
    }
  }
}

// ---------------------------------------------------------------------------
// Edge post: h = x_ji + silu(agg@Wup); residual tower; out = final h.
// Block = 64 edges, 12 waves; wave nt owns 16 cols, keeps 4 fp32 h-frags in
// VGPRs; bf16 copies ping-pong through LDS as the next layer's A matrix.
__global__ __launch_bounds__(384) void edge_post_kernel(
    const float* __restrict__ agg, const __bf16* __restrict__ Wup_p,
    const float* __restrict__ x_ji, const float* __restrict__ x,
    const __bf16* __restrict__ Wrb_p, const float* __restrict__ rbB,
    const __bf16* __restrict__ Wlin_p, const float* __restrict__ b_lin,
    const __bf16* __restrict__ Wra_p, const float* __restrict__ raB,
    float* __restrict__ out, int E) {
  __shared__ float  sG[MT * GSTR];    // TDM-padded f32 agg tile
  __shared__ __bf16 sH[MT * LSTR];
  __shared__ __bf16 sT[MT * LSTR];
  const int e0 = blockIdx.x * MT;
  const int tid = threadIdx.x, lane = tid & 31, wave = tid >> 5;
  const int nt = wave, nl = lane & 15;
  const int rbase = (lane & 16) ? 8 : 0;
  const int col = nt * 16 + nl;

  if (wave == 0) {
    tdm_load_2d_f32((unsigned)(size_t)sG, agg + (size_t)e0 * IDIM,
                    IDIM, (unsigned)E, IDIM, MT, IDIM, 1);
    tdm_wait();
  }
  __syncthreads();

  v8f h[4];
  {
    v8f acc[4] = {};
#pragma unroll
    for (int ks = 0; ks < 2; ++ks) {   // K = 64
      bf16x16 b = load_b_frag(Wup_p, ks * 12 + nt, lane);
#pragma unroll
      for (int rt = 0; rt < 4; ++rt) {
        bf16x16 a = load_a_frag_f32(sG, GSTR, rt * 16, ks * 32, lane);
        acc[rt] = wmma_bf16(a, b, acc[rt]);
      }
    }
#pragma unroll
    for (int rt = 0; rt < 4; ++rt) {
#pragma unroll
      for (int r = 0; r < 8; ++r) {
        int er = e0 + rt * 16 + rbase + r;
        float xj = (er < E) ? x_ji[(size_t)er * H + col] : 0.0f;
        h[rt][r] = silu_f(acc[rt][r]) + xj;
      }
      store_frag(sH, h[rt], rt * 16, rbase, col);
    }
  }
  __syncthreads();

  v8f t[4], t2[4];
  // residual-before (1 layer)
  layer192_m64(sH, Wrb_p, rbB, nt, lane, col, t);
#pragma unroll
  for (int rt = 0; rt < 4; ++rt) store_frag(sT, t[rt], rt * 16, rbase, col);
  __syncthreads();
  layer192_m64(sT, Wrb_p + MS, rbB + H, nt, lane, col, t2);
#pragma unroll
  for (int rt = 0; rt < 4; ++rt) {
#pragma unroll
    for (int r = 0; r < 8; ++r) h[rt][r] += t2[rt][r];
    store_frag(sH, h[rt], rt * 16, rbase, col);
  }
  __syncthreads();

  // h = silu(h@W_lin + b) + x
  layer192_m64(sH, Wlin_p, b_lin, nt, lane, col, t);
#pragma unroll
  for (int rt = 0; rt < 4; ++rt) {
#pragma unroll
    for (int r = 0; r < 8; ++r) {
      int er = e0 + rt * 16 + rbase + r;
      float xv = (er < E) ? x[(size_t)er * H + col] : 0.0f;
      h[rt][r] = t[rt][r] + xv;
    }
    store_frag(sH, h[rt], rt * 16, rbase, col);
  }
  __syncthreads();

  // residual-after (2 layers)
  for (int l = 0; l < 2; ++l) {
    layer192_m64(sH, Wra_p + (size_t)(2 * l) * MS, raB + (2 * l) * H,
                 nt, lane, col, t);
#pragma unroll
    for (int rt = 0; rt < 4; ++rt) store_frag(sT, t[rt], rt * 16, rbase, col);
    __syncthreads();
    layer192_m64(sT, Wra_p + (size_t)(2 * l + 1) * MS, raB + (2 * l + 1) * H,
                 nt, lane, col, t2);
#pragma unroll
    for (int rt = 0; rt < 4; ++rt) {
#pragma unroll
      for (int r = 0; r < 8; ++r) h[rt][r] += t2[rt][r];
      store_frag(sH, h[rt], rt * 16, rbase, col);
    }
    __syncthreads();
  }

#pragma unroll
  for (int rt = 0; rt < 4; ++rt)
#pragma unroll
    for (int r = 0; r < 8; ++r) {
      int er = e0 + rt * 16 + rbase + r;
      if (er < E) out[(size_t)er * H + col] = h[rt][r];
    }
}

// ---------------------------------------------------------------------------
extern "C" void kernel_launch(void* const* d_in, const int* in_sizes, int n_in,
                              void* d_out, int out_size, void* d_ws, size_t ws_size,
                              hipStream_t stream) {
  const float* x      = (const float*)d_in[0];
  const float* rbf    = (const float*)d_in[1];
  const float* sbf    = (const float*)d_in[2];
  const int*   idx_kj = (const int*)d_in[3];
  const int*   idx_ji = (const int*)d_in[4];
  const float* W_rbf1 = (const float*)d_in[5];
  const float* W_rbf2 = (const float*)d_in[6];
  const float* W_sbf1 = (const float*)d_in[7];
  const float* W_sbf2 = (const float*)d_in[8];
  const float* W_kj   = (const float*)d_in[9];
  const float* b_kj   = (const float*)d_in[10];
  const float* W_ji   = (const float*)d_in[11];
  const float* b_ji   = (const float*)d_in[12];
  const float* W_down = (const float*)d_in[13];
  const float* W_up   = (const float*)d_in[14];
  const float* rbW    = (const float*)d_in[15];
  const float* rbB    = (const float*)d_in[16];
  const float* W_lin  = (const float*)d_in[17];
  const float* b_lin  = (const float*)d_in[18];
  const float* raW    = (const float*)d_in[19];
  const float* raB    = (const float*)d_in[20];

  const int  E = in_sizes[0] / H;
  const long T = (long)in_sizes[2] / NSBF;

  char* w = (char*)d_ws;
  auto alloc = [&](size_t bytes) {
    char* p = w;
    w += (bytes + 255) & ~(size_t)255;
    return p;
  };
  float*  x_ji   = (float*)alloc((size_t)E * H * 4);
  float*  xkj    = (float*)alloc((size_t)E * IDIM * 4);
  float*  agg    = (float*)alloc((size_t)E * IDIM * 4);
  float*  Wrbf_c = (float*)alloc((size_t)NRAD * H * 4);
  float*  Wsbf_c = (float*)alloc((size_t)NSBF * IDIM * 4);
  __bf16* Wji_p  = (__bf16*)alloc((size_t)MS * 2);
  __bf16* Wkj_p  = (__bf16*)alloc((size_t)MS * 2);
  __bf16* Wdn_p  = (__bf16*)alloc((size_t)H * IDIM * 2);
  __bf16* Wup_p  = (__bf16*)alloc((size_t)IDIM * H * 2);
  __bf16* Wsb_p  = (__bf16*)alloc((size_t)64 * IDIM * 2);
  __bf16* Wrb_p  = (__bf16*)alloc((size_t)2 * MS * 2);
  __bf16* Wln_p  = (__bf16*)alloc((size_t)MS * 2);
  __bf16* Wra_p  = (__bf16*)alloc((size_t)4 * MS * 2);
  (void)ws_size; (void)n_in;

  {
    long n = (long)E * IDIM;
    zero_f32_kernel<<<(int)((n + 255) / 256), 256, 0, stream>>>(agg, n);
  }
  combine_weights_kernel<<<(NRAD * H + NSBF * IDIM + 255) / 256, 256, 0, stream>>>(
      W_rbf1, W_rbf2, W_sbf1, W_sbf2, Wrbf_c, Wsbf_c);
  auto pack = [&](const float* src, __bf16* dst, int Ksrc, int Kpad, int N) {
    int total = Kpad * N;
    pack_weights_kernel<<<(total + 255) / 256, 256, 0, stream>>>(src, dst, Ksrc, Kpad, N);
  };
  pack(W_ji,   Wji_p,      H,   H, H);
  pack(W_kj,   Wkj_p,      H,   H, H);
  pack(W_down, Wdn_p,      H,   H, IDIM);
  pack(W_up,   Wup_p,   IDIM, IDIM, H);
  pack(Wsbf_c, Wsb_p,   NSBF,  64, IDIM);
  pack(rbW,          Wrb_p,          H, H, H);
  pack(rbW + MS,     Wrb_p + MS,     H, H, H);
  pack(W_lin,        Wln_p,          H, H, H);
  pack(raW,          Wra_p,          H, H, H);
  pack(raW + MS,     Wra_p + MS,     H, H, H);
  pack(raW + 2 * MS, Wra_p + 2 * MS, H, H, H);
  pack(raW + 3 * MS, Wra_p + 3 * MS, H, H, H);

  const int gridE = (E + MT - 1) / MT;
  edge_pre_kernel<<<gridE, 384, 0, stream>>>(x, rbf, Wji_p, Wkj_p, Wdn_p, b_ji, b_kj,
                                             Wrbf_c, x_ji, xkj, E);
  {
    int grid = (int)((T + 255) / 256);
    triplet_kernel<<<grid, 256, 0, stream>>>(sbf, idx_kj, idx_ji, Wsb_p, xkj, agg, T);
  }
  edge_post_kernel<<<gridE, 384, 0, stream>>>(agg, Wup_p, x_ji, x, Wrb_p, rbB, Wln_p,
                                              b_lin, Wra_p, raB, (float*)d_out, E);
}
